// GINet_34660386078855
// MI455X (gfx1250) — compile-verified, matrix operand order checked by reading, not run on hardware
//
#include <hip/hip_runtime.h>
#include <hip/hip_bf16.h>

#define NN      100000
#define EE      300000
#define DD      256
#define FDIM    512
#define LL      5
#define GG      4000
#define NBOND   2
#define EPSV    1e-5f

typedef __attribute__((ext_vector_type(16))) __bf16 v16bf;
typedef __attribute__((ext_vector_type(8)))  __bf16 v8bf;
typedef __attribute__((ext_vector_type(4)))  __bf16 bf16x4;
typedef __attribute__((ext_vector_type(8)))  float  v8f;
typedef __attribute__((ext_vector_type(4)))  float  f32x4;

__device__ __forceinline__ float softplus_f(float x) {
    return x > 20.f ? x : log1pf(__expf(x));
}

// packed f32x4 -> bf16x4 conversion + single 8-byte LDS store
__device__ __forceinline__ void st_bf16x4(__bf16* p, f32x4 v) {
    *(bf16x4*)p = __builtin_convertvector(v, bf16x4);
}

// ---------------------------------------------------------------------------
// h[i,c] = x_emb1[atomics[i],c] + pos[i,:] @ x_emb2_w[:,c] + b[c]
// ---------------------------------------------------------------------------
__global__ __launch_bounds__(256)
void init_h_kernel(const int* __restrict__ atomics, const float* __restrict__ pos,
                   const float* __restrict__ emb1, const float* __restrict__ w,
                   const float* __restrict__ b, float* __restrict__ h, int n)
{
    size_t gid = (size_t)blockIdx.x * 256 + threadIdx.x;
    if (gid >= (size_t)n * DD) return;
    int i = (int)(gid >> 8);
    int c = (int)(gid & (DD - 1));
    int a = atomics[i];
    float p0 = pos[(size_t)i * 3 + 0];
    float p1 = pos[(size_t)i * 3 + 1];
    float p2 = pos[(size_t)i * 3 + 2];
    h[gid] = emb1[(size_t)a * DD + c] + p0 * w[c] + p1 * w[DD + c] + p2 * w[2 * DD + c] + b[c];
}

// ---------------------------------------------------------------------------
// agg[i,c] = h[i,c] + eemb_self[c]   (fuses zero-init + self-loop message)
// ---------------------------------------------------------------------------
__global__ __launch_bounds__(256)
void init_agg_kernel(const float* __restrict__ h, const float* __restrict__ eemb_self,
                     float* __restrict__ agg, size_t total)
{
    size_t gid = (size_t)blockIdx.x * 256 + threadIdx.x;
    if (gid >= total) return;
    int c = (int)(gid & (DD - 1));
    agg[gid] = h[gid] + eemb_self[c];
}

// ---------------------------------------------------------------------------
// scatter-add real edges: agg[dst[e],:] += h[src[e],:] + edge_emb[eattr[e],:]
// ---------------------------------------------------------------------------
__global__ __launch_bounds__(256)
void scatter_kernel(const float* __restrict__ h, const int* __restrict__ src,
                    const int* __restrict__ dst, const int* __restrict__ eattr,
                    const float* __restrict__ eemb, float* __restrict__ agg, int nedges)
{
    size_t gid = (size_t)blockIdx.x * 256 + threadIdx.x;
    int e  = (int)(gid >> 6);
    int c  = (int)(gid & 63) << 2;
    if (e >= nedges) return;
    int s = src[e], d = dst[e], a = eattr[e];
    float4 hv = *(const float4*)(h + (size_t)s * DD + c);
    float4 ev = *(const float4*)(eemb + (size_t)a * DD + c);
    float* o = agg + (size_t)d * DD + c;
    atomicAdd(o + 0, hv.x + ev.x);
    atomicAdd(o + 1, hv.y + ev.y);
    atomicAdd(o + 2, hv.z + ev.z);
    atomicAdd(o + 3, hv.w + ev.w);
}

// ---------------------------------------------------------------------------
// WMMA GEMM:  Y[M,Nout] = act(X[M,K] @ W[K,Nout] + bias)
// block tile 128x64, BK=64, 8 waves, each wave a 16x64 strip.
// Software pipeline: global->reg prefetch of tile k+1 overlaps the 8 WMMAs
// of tile k. Row index clamped (not branched) so loads issue unconditionally.
// ---------------------------------------------------------------------------
#define BK 64

__global__ __launch_bounds__(256)
void wmma_gemm_bias_act(const float* __restrict__ X, const float* __restrict__ W,
                        const float* __restrict__ bias, float* __restrict__ Y,
                        int M, int K, int Nout, int act)
{
    __shared__ __align__(16) __bf16 As[128][BK];   // 16 KB
    __shared__ __align__(16) __bf16 Bt[64][BK];    //  8 KB (transposed: [n][k])

    const int tid  = threadIdx.x;
    const int wave = tid >> 5;
    const int lane = tid & 31;
    const int half = lane >> 4;
    const int lm   = lane & 15;
    const int rowBase = blockIdx.x * 128;
    const int colBase = blockIdx.y * 64;

    // --- staging coordinates (fixed per thread) ---
    // A: 2048 float4 groups: r = g>>4 (row), kq = (g&15)*4
    // B: 1024 dwordx4 groups: n = g&63, kq = (g>>6)*4 (4 strided dword loads,
    //    W is small & L2-resident; enables 8-byte transposed LDS store)
    int ar[8], akq[8];
    #pragma unroll
    for (int it = 0; it < 8; ++it) {
        int g = tid + it * 256;
        int r = g >> 4;
        ar[it]  = (rowBase + r < M) ? (rowBase + r) : (M - 1);   // clamp, no branch
        akq[it] = (g & 15) << 2;
    }
    int bn[4], bkq[4];
    #pragma unroll
    for (int it = 0; it < 4; ++it) {
        int g = tid + it * 256;
        bn[it]  = g & 63;
        bkq[it] = (g >> 6) << 2;
    }

    f32x4 areg[8];
    f32x4 breg[4];

    auto ldg = [&](int k0) {
        #pragma unroll
        for (int it = 0; it < 8; ++it)
            areg[it] = *(const f32x4*)(X + (size_t)ar[it] * K + k0 + akq[it]);
        #pragma unroll
        for (int it = 0; it < 4; ++it) {
            const float* wp = W + (size_t)(k0 + bkq[it]) * Nout + colBase + bn[it];
            f32x4 v;
            v.x = wp[0];
            v.y = wp[(size_t)Nout];
            v.z = wp[(size_t)2 * Nout];
            v.w = wp[(size_t)3 * Nout];
            breg[it] = v;
        }
    };
    auto sts = [&]() {
        #pragma unroll
        for (int it = 0; it < 8; ++it)
            st_bf16x4(&As[(tid + it * 256) >> 4][akq[it]], areg[it]);
        #pragma unroll
        for (int it = 0; it < 4; ++it)
            st_bf16x4(&Bt[bn[it]][bkq[it]], breg[it]);
    };

    v8f c[4] = {};

    ldg(0);
    for (int k0 = 0; k0 < K; k0 += BK) {
        sts();
        __syncthreads();
        // prefetch next tile (clamped on last iter; result unused)
        ldg((k0 + BK < K) ? (k0 + BK) : k0);

        const int arow = wave * 16 + lm;
        #pragma unroll
        for (int kb = 0; kb < BK; kb += 32) {
            // A fragment: lane halves hold K {0..7,16..23} / {8..15,24..31}
            union { v16bf v; v8bf h[2]; } af;
            af.h[0] = *(const v8bf*)(&As[arow][kb + half * 8]);
            af.h[1] = *(const v8bf*)(&As[arow][kb + 16 + half * 8]);
            union { v16bf v; v8bf h[2]; } bf[4];
            #pragma unroll
            for (int t = 0; t < 4; ++t) {
                const int bcol = t * 16 + lm;
                bf[t].h[0] = *(const v8bf*)(&Bt[bcol][kb + half * 8]);
                bf[t].h[1] = *(const v8bf*)(&Bt[bcol][kb + 16 + half * 8]);
            }
            #pragma unroll
            for (int t = 0; t < 4; ++t)
                c[t] = __builtin_amdgcn_wmma_f32_16x16x32_bf16(
                           false, af.v, false, bf[t].v, (short)0, c[t], false, false);
        }
        __syncthreads();
    }

    // ---- epilogue: bias + optional softplus, f32 store
    #pragma unroll
    for (int t = 0; t < 4; ++t) {
        union { v8f v; float f[8]; } r;
        r.v = c[t];
        const int col = colBase + t * 16 + lm;
        const float bv = bias[col];
        #pragma unroll
        for (int i = 0; i < 8; ++i) {
            int row = rowBase + wave * 16 + half * 8 + i;
            if (row < M) {
                float val = r.f[i] + bv;
                if (act) val = softplus_f(val);
                Y[(size_t)row * Nout + col] = val;
            }
        }
    }
}

// ---------------------------------------------------------------------------
// BatchNorm column statistics: per-block register accumulation, one atomic
// per column per block.
// ---------------------------------------------------------------------------
__global__ __launch_bounds__(256)
void bn_stats_kernel(const float* __restrict__ h, float* __restrict__ sum,
                     float* __restrict__ sumsq, int nrows)
{
    int d = threadIdx.x;                    // 256 columns
    float s = 0.f, q = 0.f;
    for (int i = blockIdx.x; i < nrows; i += gridDim.x) {
        float v = h[(size_t)i * DD + d];
        s += v;
        q += v * v;
    }
    atomicAdd(&sum[d], s);
    atomicAdd(&sumsq[d], q);
}

__global__ __launch_bounds__(256)
void bn_apply_kernel(float* __restrict__ h, const float* __restrict__ sum,
                     const float* __restrict__ sumsq, const float* __restrict__ gamma,
                     const float* __restrict__ beta, int nrows, int act)
{
    size_t gid = (size_t)blockIdx.x * 256 + threadIdx.x;
    if (gid >= (size_t)nrows * DD) return;
    int d = (int)(gid & (DD - 1));
    float inv  = 1.f / (float)nrows;
    float mean = sum[d] * inv;
    float var  = sumsq[d] * inv - mean * mean;
    float v = (h[gid] - mean) * rsqrtf(var + EPSV) * gamma[d] + beta[d];
    if (act) v = softplus_f(v);
    h[gid] = v;
}

// ---------------------------------------------------------------------------
// helpers: zero, graph counts, mean pool
// ---------------------------------------------------------------------------
__global__ __launch_bounds__(256)
void zero_kernel(float* __restrict__ p, size_t n)
{
    size_t gid = (size_t)blockIdx.x * 256 + threadIdx.x;
    if (gid < n) p[gid] = 0.f;
}

__global__ __launch_bounds__(256)
void count_kernel(const int* __restrict__ batch, float* __restrict__ cnt, int n)
{
    size_t gid = (size_t)blockIdx.x * 256 + threadIdx.x;
    if (gid >= (size_t)n) return;
    atomicAdd(&cnt[batch[gid]], 1.f);
}

__global__ __launch_bounds__(256)
void pool_kernel(const float* __restrict__ feat, const int* __restrict__ batch,
                 float* __restrict__ gsum, int n)
{
    size_t gid = (size_t)blockIdx.x * 256 + threadIdx.x;
    if (gid >= (size_t)n * (FDIM / 4)) return;
    int i = (int)(gid / (FDIM / 4));
    int c = (int)(gid % (FDIM / 4)) << 2;
    int g = batch[i];
    float4 v = *(const float4*)(feat + (size_t)i * FDIM + c);
    float* o = gsum + (size_t)g * FDIM + c;
    atomicAdd(o + 0, v.x);
    atomicAdd(o + 1, v.y);
    atomicAdd(o + 2, v.z);
    atomicAdd(o + 3, v.w);
}

__global__ __launch_bounds__(256)
void gmean_kernel(const float* __restrict__ gsum, const float* __restrict__ cnt,
                  float* __restrict__ gfeat, size_t total)
{
    size_t gid = (size_t)blockIdx.x * 256 + threadIdx.x;
    if (gid >= total) return;
    int g = (int)(gid >> 9);                // FDIM == 512
    gfeat[gid] = gsum[gid] / fmaxf(cnt[g], 1.f);
}

// final: out[g] = ghid[g,:] . head_w2 + b2     (256-wide dot per graph)
__global__ __launch_bounds__(256)
void head_out_kernel(const float* __restrict__ ghid, const float* __restrict__ w2,
                     const float* __restrict__ b2, float* __restrict__ out)
{
    __shared__ float red[256];
    int g = blockIdx.x;
    int t = threadIdx.x;
    red[t] = ghid[(size_t)g * (FDIM / 2) + t] * w2[t];
    __syncthreads();
    for (int s = 128; s > 0; s >>= 1) {
        if (t < s) red[t] += red[t + s];
        __syncthreads();
    }
    if (t == 0) out[g] = red[0] + b2[0];
}

// ---------------------------------------------------------------------------
extern "C" void kernel_launch(void* const* d_in, const int* in_sizes, int n_in,
                              void* d_out, int out_size, void* d_ws, size_t ws_size,
                              hipStream_t stream)
{
    (void)in_sizes; (void)n_in; (void)out_size; (void)ws_size;

    const int*   atomics = (const int*)  d_in[0];
    const float* pos     = (const float*)d_in[1];
    const int*   eidx    = (const int*)  d_in[2];
    const int*   eattr   = (const int*)  d_in[3];
    const int*   batch   = (const int*)  d_in[4];
    const float* x_emb1  = (const float*)d_in[5];
    const float* x_emb2w = (const float*)d_in[6];
    const float* x_emb2b = (const float*)d_in[7];
    const float* edgeemb = (const float*)d_in[8];    // [L, NBOND, D]
    const float* mlp_w1  = (const float*)d_in[9];    // [L, D, 2D]
    const float* mlp_b1  = (const float*)d_in[10];   // [L, 2D]
    const float* mlp_w2  = (const float*)d_in[11];   // [L, 2D, D]
    const float* mlp_b2  = (const float*)d_in[12];   // [L, D]
    const float* bn_g    = (const float*)d_in[13];   // [L, D]
    const float* bn_b    = (const float*)d_in[14];   // [L, D]
    const float* feat_w  = (const float*)d_in[15];   // [D, FD]
    const float* feat_b  = (const float*)d_in[16];   // [FD]
    const float* head_w1 = (const float*)d_in[17];   // [FD, FD/2]
    const float* head_b1 = (const float*)d_in[18];   // [FD/2]
    const float* head_w2 = (const float*)d_in[19];   // [FD/2, 1]
    const float* head_b2 = (const float*)d_in[20];   // [1]
    float*       out     = (float*)d_out;

    const int* src = eidx;           // edge_index[0]
    const int* dst = eidx + EE;      // edge_index[1]

    // ---- workspace layout (floats) ----
    float* h     = (float*)d_ws;                         // N*D
    float* agg   = h     + (size_t)NN * DD;              // N*D
    float* hid   = agg   + (size_t)NN * DD;              // N*2D (also feat buffer)
    float* gsum  = hid   + (size_t)NN * 2 * DD;          // G*FD
    float* cnt   = gsum  + (size_t)GG * FDIM;            // G
    float* gfeat = cnt   + (size_t)GG;                   // G*FD
    float* ghid  = gfeat + (size_t)GG * FDIM;            // G*FD/2
    float* bnsum = ghid  + (size_t)GG * (FDIM / 2);      // D
    float* bnsq  = bnsum + DD;                           // D

    const int elemBlocksND = (int)(((size_t)NN * DD + 255) / 256);

    // 1) initial node features
    init_h_kernel<<<elemBlocksND, 256, 0, stream>>>(atomics, pos, x_emb1, x_emb2w,
                                                    x_emb2b, h, NN);

    // 2) message-passing layers
    for (int l = 0; l < LL; ++l) {
        const float* eemb = edgeemb + (size_t)l * NBOND * DD;
        // agg = h + self-loop embedding (zero-init fused)
        init_agg_kernel<<<elemBlocksND, 256, 0, stream>>>(
            h, eemb + (size_t)(NBOND - 1) * DD, agg, (size_t)NN * DD);
        // scatter real edges
        {
            int blocks = (int)(((size_t)EE * 64 + 255) / 256);
            scatter_kernel<<<blocks, 256, 0, stream>>>(h, src, dst, eattr, eemb, agg, EE);
        }
        // hid = softplus(agg @ W1 + b1)    [N,512]
        {
            dim3 grid((NN + 127) / 128, (2 * DD) / 64);
            wmma_gemm_bias_act<<<grid, 256, 0, stream>>>(
                agg, mlp_w1 + (size_t)l * DD * 2 * DD, mlp_b1 + (size_t)l * 2 * DD,
                hid, NN, DD, 2 * DD, 1);
        }
        // h = hid @ W2 + b2                [N,256]
        {
            dim3 grid((NN + 127) / 128, DD / 64);
            wmma_gemm_bias_act<<<grid, 256, 0, stream>>>(
                hid, mlp_w2 + (size_t)l * 2 * DD * DD, mlp_b2 + (size_t)l * DD,
                h, NN, 2 * DD, DD, 0);
        }
        // batchnorm (+ softplus except last layer)
        zero_kernel<<<2, 256, 0, stream>>>(bnsum, 2 * DD);   // bnsum & bnsq contiguous
        bn_stats_kernel<<<512, 256, 0, stream>>>(h, bnsum, bnsq, NN);
        bn_apply_kernel<<<elemBlocksND, 256, 0, stream>>>(
            h, bnsum, bnsq, bn_g + (size_t)l * DD, bn_b + (size_t)l * DD,
            NN, (l < LL - 1) ? 1 : 0);
    }

    // 3) feat = h @ feat_w + feat_b        [N,512] (reuse hid buffer)
    {
        dim3 grid((NN + 127) / 128, FDIM / 64);
        wmma_gemm_bias_act<<<grid, 256, 0, stream>>>(h, feat_w, feat_b, hid,
                                                     NN, DD, FDIM, 0);
    }

    // 4) global mean pool
    {
        size_t nz = (size_t)GG * FDIM + GG;               // gsum + cnt contiguous
        zero_kernel<<<(int)((nz + 255) / 256), 256, 0, stream>>>(gsum, nz);
        count_kernel<<<(NN + 255) / 256, 256, 0, stream>>>(batch, cnt, NN);
        int blocks = (int)(((size_t)NN * (FDIM / 4) + 255) / 256);
        pool_kernel<<<blocks, 256, 0, stream>>>(hid, batch, gsum, NN);
        size_t tg = (size_t)GG * FDIM;
        gmean_kernel<<<(int)((tg + 255) / 256), 256, 0, stream>>>(gsum, cnt, gfeat, tg);
    }

    // 5) head: ghid = softplus(gfeat @ W1 + b1); out = ghid @ w2 + b2
    {
        dim3 grid((GG + 127) / 128, (FDIM / 2) / 64);
        wmma_gemm_bias_act<<<grid, 256, 0, stream>>>(gfeat, head_w1, head_b1, ghid,
                                                     GG, FDIM, FDIM / 2, 1);
        head_out_kernel<<<GG, 256, 0, stream>>>(ghid, head_w2, head_b2, out);
    }
}
